// LBKTcell_83975200571953
// MI455X (gfx1250) — compile-verified, matrix operand order checked by reading, not run on hardware
//
#include <hip/hip_runtime.h>
#include <math.h>

typedef _Float16 half_t;
typedef __attribute__((ext_vector_type(16))) _Float16 v16h;
typedef __attribute__((ext_vector_type(8)))  float    v8f;
typedef __attribute__((ext_vector_type(4)))  float    v4f;
typedef __attribute__((ext_vector_type(4)))  int      v4i;

#define NUK 128
#define MSK 64
#define BATCH 4096
#define INROW 323   // NU+MS+DT+3
#define D_F 10.0f
#define K_F 0.3f
#define B_F 0.3f

#if defined(__gfx1250__) && __has_builtin(__builtin_amdgcn_global_load_async_to_lds_b128)
#define HAVE_ASYNC_LDS 1
#endif

__device__ __forceinline__ float sigmoidf(float x){ return 1.0f/(1.0f+__expf(-x)); }

__device__ __forceinline__ v8f vzero8(){
  v8f z;
#pragma unroll
  for(int i=0;i<8;i++) z[i]=0.0f;
  return z;
}

__device__ __forceinline__ v8f wmma16(v16h a, v16h b, v8f c){
  // D = A(16x32 f16) x B(32x16 f16) + C(16x16 f32)
  return __builtin_amdgcn_wmma_f32_16x16x32_f16(false, a, false, b, (short)0, c, false, false);
}

// packed B tiles: tile = kt*8 + nt, 512 halves per tile, lane-major (lane*16+h)
__device__ __forceinline__ v16h load_btile(const half_t* pk, int tile, int lane){
  return *(const v16h*)(pk + (size_t)tile*512 + (size_t)lane*16);
}

// ISA A/B fragment K index for half h in lane
__device__ __forceinline__ int amap_k(int h, int lane){ return h + 8*((h>>3) + (lane>>4)); }

// ---------------- weight packing (once per launch, tiny) ----------------
__device__ void pack_w(const float* W, int Ktiles, half_t* dst, int tid, int nth){
  int total = Ktiles*8*32;
  for (int idx = tid; idx < total; idx += nth){
    int t = idx >> 5, lane = idx & 31;
    int kt = t >> 3, nt = t & 7;
    half_t* d = dst + (size_t)t*512 + (size_t)lane*16;
    int n = nt*16 + (lane & 15);
#pragma unroll
    for (int h=0; h<16; h++){
      int k = kt*32 + amap_k(h, lane);
      d[h] = (half_t)W[(size_t)k*NUK + n];
    }
  }
}

__global__ void __launch_bounds__(256) k_prep(
    const float* g3W, const float* outW,
    const float* tgW, const float* agW, const float* hgW,
    const float* tW, const float* aW, const float* hW,
    half_t* W1pk, half_t* W2pk, half_t* outWpk,
    half_t* tgWpk, half_t* agWpk, half_t* hgWpk,
    half_t* TWpk, half_t* AWpk, half_t* HWpk,
    float* csum)
{
  int tid = threadIdx.x, nth = blockDim.x;
  for (int n = tid; n < NUK; n += nth){
    float ct=0.f, ca=0.f, ch=0.f;
    for (int k=0;k<50;k++){
      ct += g3W[(size_t)(256+k)*NUK+n];
      ca += g3W[(size_t)(306+k)*NUK+n];
      ch += g3W[(size_t)(356+k)*NUK+n];
    }
    csum[n]=ct; csum[NUK+n]=ca; csum[2*NUK+n]=ch;
  }
  pack_w(g3W,          4, W1pk,   tid, nth);   // h_pre rows 0..127
  pack_w(g3W+128*NUK,  4, W2pk,   tid, nth);   // interact rows 128..255
  pack_w(outW,         8, outWpk, tid, nth);
  pack_w(tgW,          8, tgWpk,  tid, nth);
  pack_w(agW,          8, agWpk,  tid, nth);
  pack_w(hgW,          8, hgWpk,  tid, nth);
  for (int r=0;r<4;r++){
    pack_w(tW + (size_t)r*129*NUK, 4, TWpk + (size_t)r*16384, tid, nth);
    pack_w(aW + (size_t)r*129*NUK, 4, AWpk + (size_t)r*16384, tid, nth);
    pack_w(hW + (size_t)r*129*NUK, 4, HWpk + (size_t)r*16384, tid, nth);
  }
}

// ---------------- h_pre_tilde[b,n] = sum_m corr[b,m]*h_pre[b,m,n] ----------------
__global__ void __launch_bounds__(128) k_tilde(const float* inputs, const float* states, float* hpt){
  int b = blockIdx.x, n = threadIdx.x;
  const float* hp   = states + (size_t)b*MSK*NUK;
  const float* corr = inputs + (size_t)b*INROW + 128;
  float s = 0.f;
#pragma unroll 4
  for (int m=0; m<MSK; m++) s += corr[m]*hp[(size_t)m*NUK + n];
  hpt[(size_t)b*NUK + n] = s;
}

// ---------------- preds / after_preds: sigmoid([src, topic] @ out_W + b).mean ----------------
__global__ void __launch_bounds__(128) k_predict(
    const float* src, const float* inputs, const half_t* outWpk, const float* outb,
    float* preds, const float* preds_prev, float* result, int mode)
{
  int lane = threadIdx.x & 31, w = threadIdx.x >> 5;
  int row0 = blockIdx.x*64 + w*16;
  v8f acc[8];
#pragma unroll
  for (int i=0;i<8;i++) acc[i]=vzero8();
  int arow = row0 + (lane & 15);
  const float* s1 = src + (size_t)arow*NUK;
  const float* s2 = inputs + (size_t)arow*INROW + 192; // topic
  for (int kt=0; kt<8; kt++){
    const float* sp = (kt < 4) ? s1 : (s2 - 128);
    v16h af;
#pragma unroll
    for (int h=0; h<16; h++) af[h] = (half_t)sp[kt*32 + amap_k(h,lane)];
#pragma unroll
    for (int nt=0; nt<8; nt++)
      acc[nt] = wmma16(af, load_btile(outWpk, kt*8+nt, lane), acc[nt]);
  }
  float rowsum[8];
#pragma unroll
  for (int i=0;i<8;i++) rowsum[i]=0.f;
#pragma unroll
  for (int nt=0; nt<8; nt++){
    float bc = outb[nt*16 + (lane&15)];
#pragma unroll
    for (int i=0;i<8;i++) rowsum[i] += sigmoidf(acc[nt][i] + bc);
  }
#pragma unroll
  for (int m=8; m>=1; m>>=1){
#pragma unroll
    for (int i=0;i<8;i++) rowsum[i] += __shfl_xor(rowsum[i], m, 32);
  }
  if ((lane & 15) == 0){
    int half8 = (lane >> 4) * 8;
    for (int i=0;i<8;i++){
      int row = row0 + half8 + i;
      float p = rowsum[i] * (1.0f/128.0f);
      if (mode == 0) preds[row] = p;
      else {
        float p0 = preds_prev[row];
        result[(size_t)row*2+0] = p0;
        result[(size_t)row*2+1] = (p - p0) / (1.0f - p0);
      }
    }
  }
}

// ---------------- gains (3x sigmoid GEMM) + bvec (interact@W2 + scalar colsums + b3) ----------------
__global__ void __launch_bounds__(128) k_gains(
    const float* inputs, const float* hpt,
    const half_t* tgWpk, const half_t* agWpk, const half_t* hgWpk, const half_t* W2pk,
    const float* tgb, const float* agb, const float* hgb,
    const float* csum, const float* g3b,
    float* gT, float* gA, float* gH, float* bvec)
{
  int lane = threadIdx.x & 31, w = threadIdx.x >> 5;
  int row0 = blockIdx.x*32 + (w & 1)*16;
  int col0 = (w >> 1)*64;
  int ntb  = col0 >> 4;
  v8f aT[4],aA[4],aH[4],aV[4];
#pragma unroll
  for (int i=0;i<4;i++){ aT[i]=vzero8(); aA[i]=vzero8(); aH[i]=vzero8(); aV[i]=vzero8(); }
  int arow = row0 + (lane & 15);
  const float* s1 = hpt + (size_t)arow*NUK;
  const float* s2 = inputs + (size_t)arow*INROW; // interact at col 0
  for (int kt=0; kt<8; kt++){
    const float* sp = (kt < 4) ? s1 : (s2 - 128);
    v16h af;
#pragma unroll
    for (int h=0; h<16; h++) af[h] = (half_t)sp[kt*32 + amap_k(h,lane)];
#pragma unroll
    for (int nt=0; nt<4; nt++){
      int tile = kt*8 + ntb + nt;
      aT[nt] = wmma16(af, load_btile(tgWpk, tile, lane), aT[nt]);
      aA[nt] = wmma16(af, load_btile(agWpk, tile, lane), aA[nt]);
      aH[nt] = wmma16(af, load_btile(hgWpk, tile, lane), aH[nt]);
    }
    if (kt >= 4){ // af is pure interact here -> reuse for W2
#pragma unroll
      for (int nt=0; nt<4; nt++)
        aV[nt] = wmma16(af, load_btile(W2pk, (kt-4)*8 + ntb + nt, lane), aV[nt]);
    }
  }
  int half8 = (lane >> 4) * 8;
  for (int i=0;i<8;i++){
    int row = row0 + half8 + i;
    const float* inr = inputs + (size_t)row*INROW;
    float tf = inr[320], afc = inr[321], hf = inr[322];
    float gt = K_F + (1.f-K_F)*sigmoidf(D_F*(tf  - B_F));
    float ga = K_F + (1.f-K_F)*sigmoidf(D_F*(afc - B_F));
    float gh = K_F + (1.f-K_F)*sigmoidf(D_F*(hf  - B_F));
#pragma unroll
    for (int nt=0; nt<4; nt++){
      int c = col0 + nt*16 + (lane&15);
      size_t o = (size_t)row*NUK + c;
      gT[o] = sigmoidf((aT[nt][i] + tgb[c]) * gt);
      gA[o] = sigmoidf((aA[nt][i] + agb[c]) * ga);
      gH[o] = sigmoidf((aH[nt][i] + hgb[c]) * gh);
      bvec[o] = aV[nt][i] + tf*csum[c] + afc*csum[NUK+c] + hf*csum[2*NUK+c] + g3b[c];
    }
  }
}

// ---------------- fusion: learning_gain = relu(sum_r Wf[r]*prod_f(g_f@W_f[r,:128]+W_f[r,128]) + bias) ----
__global__ void __launch_bounds__(128) k_fuse(
    const float* gT, const float* gA, const float* gH,
    const half_t* TWpk, const half_t* AWpk, const half_t* HWpk,
    const float* tW, const float* aW, const float* hW,
    const float* Wf, const float* bias, float* lg)
{
  int lane = threadIdx.x & 31, w = threadIdx.x >> 5;
  int row0 = blockIdx.x*16;
  int ntb  = w*2;               // cols [32w, 32w+32)
  int arow = row0 + (lane & 15);
  v16h fT[4], fA[4], fH[4];
  for (int kt=0; kt<4; kt++){
#pragma unroll
    for (int h=0; h<16; h++){
      int k = kt*32 + amap_k(h,lane);
      fT[kt][h] = (half_t)gT[(size_t)arow*NUK + k];
      fA[kt][h] = (half_t)gA[(size_t)arow*NUK + k];
      fH[kt][h] = (half_t)gH[(size_t)arow*NUK + k];
    }
  }
  v8f res[2]; res[0]=vzero8(); res[1]=vzero8();
  for (int r=0; r<4; r++){
    v8f cT[2],cA[2],cH[2];
#pragma unroll
    for (int i=0;i<2;i++){ cT[i]=vzero8(); cA[i]=vzero8(); cH[i]=vzero8(); }
    const half_t* Tp = TWpk + (size_t)r*16384;
    const half_t* Ap = AWpk + (size_t)r*16384;
    const half_t* Hp = HWpk + (size_t)r*16384;
    for (int kt=0; kt<4; kt++){
#pragma unroll
      for (int nt=0; nt<2; nt++){
        int tile = kt*8 + ntb + nt;
        cT[nt] = wmma16(fT[kt], load_btile(Tp, tile, lane), cT[nt]);
        cA[nt] = wmma16(fA[kt], load_btile(Ap, tile, lane), cA[nt]);
        cH[nt] = wmma16(fH[kt], load_btile(Hp, tile, lane), cH[nt]);
      }
    }
    float wfr = Wf[r];
#pragma unroll
    for (int nt=0; nt<2; nt++){
      int c = ntb*16 + nt*16 + (lane&15);
      float pT = tW[((size_t)r*129 + 128)*NUK + c];
      float pA = aW[((size_t)r*129 + 128)*NUK + c];
      float pH = hW[((size_t)r*129 + 128)*NUK + c];
#pragma unroll
      for (int i=0;i<8;i++)
        res[nt][i] += wfr * (cT[nt][i]+pT)*(cA[nt][i]+pA)*(cH[nt][i]+pH);
    }
  }
  int half8 = (lane >> 4) * 8;
#pragma unroll
  for (int nt=0; nt<2; nt++){
    int c = ntb*16 + nt*16 + (lane&15);
    float bc = bias[c];
    for (int i=0;i<8;i++){
      int row = row0 + half8 + i;
      float v = res[nt][i] + bc;
      lg[(size_t)row*NUK + c] = v > 0.f ? v : 0.f;
    }
  }
}

// ---------------- main cell: gate = sigmoid(h_pre@W1 + bvec); h = h_pre*gate + corr*lg ----------------
__global__ void __launch_bounds__(128) k_cell(
    const float* inputs, const float* states,
    const half_t* W1pk, const float* bvec, const float* lg,
    float* h_out, float* htil)
{
  __shared__ float s_h[MSK*132];   // padded stride 132 (bank-shift 4)
  __shared__ float s_ht[NUK];
  __shared__ float s_corr[MSK];
  int tid = threadIdx.x, lane = tid & 31, w = tid >> 5;
  int b = blockIdx.x;
  const float* hp = states + (size_t)b*MSK*NUK;
#if defined(HAVE_ASYNC_LDS)
  // async DMA global -> LDS (ASYNCcnt path), no VGPR staging
  for (int f4 = tid; f4 < (MSK*NUK)/4; f4 += 128){
    int f = f4*4, r = f >> 7, c = f & 127;
    __builtin_amdgcn_global_load_async_to_lds_b128(
        (__attribute__((address_space(1))) v4i*)(hp + f),
        (__attribute__((address_space(3))) v4i*)&s_h[r*132 + c],
        0, 0);
  }
  asm volatile("s_wait_asynccnt 0x0" ::: "memory");
#else
  for (int f4 = tid; f4 < (MSK*NUK)/4; f4 += 128){
    int f = f4*4, r = f >> 7, c = f & 127;
    *(v4f*)(&s_h[r*132 + c]) = *(const v4f*)(hp + f);
  }
#endif
  if (tid < MSK) s_corr[tid] = inputs[(size_t)b*INROW + 128 + tid];
  s_ht[tid] = 0.f;
  __syncthreads();

  int m0 = w*16;
  v8f acc[8];
#pragma unroll
  for (int i=0;i<8;i++) acc[i]=vzero8();
  for (int kt=0; kt<4; kt++){
    int c0 = kt*32 + ((lane>>4)<<3);
    const float* rp = &s_h[(m0 + (lane&15))*132 + c0];
    v4f q0 = *(const v4f*)rp,      q1 = *(const v4f*)(rp+4);
    v4f q2 = *(const v4f*)(rp+16), q3 = *(const v4f*)(rp+20);
    v16h af;
#pragma unroll
    for (int j=0;j<4;j++){
      af[j]=(half_t)q0[j]; af[4+j]=(half_t)q1[j];
      af[8+j]=(half_t)q2[j]; af[12+j]=(half_t)q3[j];
    }
    __builtin_prefetch(W1pk + (size_t)((kt+1)&3)*8*512, 0, 1);
#pragma unroll
    for (int nt=0; nt<8; nt++)
      acc[nt] = wmma16(af, load_btile(W1pk, kt*8+nt, lane), acc[nt]);
  }

  int half8 = (lane >> 4) * 8;
  const float* bv  = bvec + (size_t)b*NUK;
  const float* lgb = lg   + (size_t)b*NUK;
#pragma unroll
  for (int nt=0; nt<8; nt++){
    int c = nt*16 + (lane&15);
    float bvc = bv[c], lgc = lgb[c];
    float part = 0.f;
#pragma unroll
    for (int i=0;i<8;i++){
      int m = m0 + half8 + i;
      float fg = sigmoidf(acc[nt][i] + bvc);
      float hv = s_h[m*132 + c]*fg + s_corr[m]*lgc;
      h_out[((size_t)b*MSK + m)*NUK + c] = hv;
      part += s_corr[m]*hv;
    }
    part += __shfl_xor(part, 16, 32);   // fold lanes 16..31 (same col, rows+8)
    if (lane < 16) atomicAdd(&s_ht[c], part);
  }
  __syncthreads();
  htil[(size_t)b*NUK + tid] = s_ht[tid];
}

// ---------------- launch ----------------
extern "C" void kernel_launch(void* const* d_in, const int* in_sizes, int n_in,
                              void* d_out, int out_size, void* d_ws, size_t ws_size,
                              hipStream_t stream) {
  (void)in_sizes; (void)n_in; (void)out_size; (void)ws_size;
  const float* inputs = (const float*)d_in[0];
  const float* states = (const float*)d_in[1];
  const float* tW     = (const float*)d_in[2];
  const float* aW     = (const float*)d_in[3];
  const float* hW     = (const float*)d_in[4];
  const float* Wf     = (const float*)d_in[5];
  const float* bias   = (const float*)d_in[6];
  const float* g3W    = (const float*)d_in[7];
  const float* g3b    = (const float*)d_in[8];
  const float* outW   = (const float*)d_in[9];
  const float* outb   = (const float*)d_in[10];
  const float* tgW    = (const float*)d_in[11];
  const float* tgb    = (const float*)d_in[12];
  const float* agW    = (const float*)d_in[13];
  const float* agb    = (const float*)d_in[14];
  const float* hgW    = (const float*)d_in[15];
  const float* hgb    = (const float*)d_in[16];
  float* out = (float*)d_out;

  char* ws = (char*)d_ws;
  size_t off = 0;
  auto alloc = [&](size_t bytes) -> char* {
    char* p = ws + off;
    off += (bytes + 255) & ~(size_t)255;
    return p;
  };
  half_t* W1pk  = (half_t*)alloc(16384*sizeof(half_t));
  half_t* W2pk  = (half_t*)alloc(16384*sizeof(half_t));
  half_t* outWpk= (half_t*)alloc(32768*sizeof(half_t));
  half_t* tgWpk = (half_t*)alloc(32768*sizeof(half_t));
  half_t* agWpk = (half_t*)alloc(32768*sizeof(half_t));
  half_t* hgWpk = (half_t*)alloc(32768*sizeof(half_t));
  half_t* TWpk  = (half_t*)alloc(4*16384*sizeof(half_t));
  half_t* AWpk  = (half_t*)alloc(4*16384*sizeof(half_t));
  half_t* HWpk  = (half_t*)alloc(4*16384*sizeof(half_t));
  float* csum   = (float*)alloc(3*NUK*sizeof(float));
  float* hpt    = (float*)alloc((size_t)BATCH*NUK*sizeof(float));
  float* gT     = (float*)alloc((size_t)BATCH*NUK*sizeof(float));
  float* gA     = (float*)alloc((size_t)BATCH*NUK*sizeof(float));
  float* gH     = (float*)alloc((size_t)BATCH*NUK*sizeof(float));
  float* bvec   = (float*)alloc((size_t)BATCH*NUK*sizeof(float));
  float* lg     = (float*)alloc((size_t)BATCH*NUK*sizeof(float));
  float* htil   = (float*)alloc((size_t)BATCH*NUK*sizeof(float));
  float* preds  = (float*)alloc((size_t)BATCH*sizeof(float));

  float* result = out;                    // (B,2)
  float* h_out  = out + (size_t)BATCH*2;  // (B,64,128)

  k_prep<<<1, 256, 0, stream>>>(g3W, outW, tgW, agW, hgW, tW, aW, hW,
                                W1pk, W2pk, outWpk, tgWpk, agWpk, hgWpk,
                                TWpk, AWpk, HWpk, csum);
  k_tilde<<<BATCH, 128, 0, stream>>>(inputs, states, hpt);
  k_predict<<<BATCH/64, 128, 0, stream>>>(hpt, inputs, outWpk, outb,
                                          preds, preds, result, 0);
  k_gains<<<BATCH/32, 128, 0, stream>>>(inputs, hpt, tgWpk, agWpk, hgWpk, W2pk,
                                        tgb, agb, hgb, csum, g3b, gT, gA, gH, bvec);
  k_fuse<<<BATCH/16, 128, 0, stream>>>(gT, gA, gH, TWpk, AWpk, HWpk,
                                       tW, aW, hW, Wf, bias, lg);
  k_cell<<<BATCH, 128, 0, stream>>>(inputs, states, W1pk, bvec, lg, h_out, htil);
  k_predict<<<BATCH/64, 128, 0, stream>>>(htil, inputs, outWpk, outb,
                                          preds, preds, result, 1);
}
